// DeformableCaptioner_18786186952902
// MI455X (gfx1250) — matrix operand order repaired and checked by reading, not compiled, not run on hardware
//
#include <hip/hip_runtime.h>
#include <hip/hip_bf16.h>
#include <math.h>

// ---------------- types ----------------
typedef __bf16 bf16_t;
typedef __attribute__((ext_vector_type(16))) __bf16 v16bf;
typedef __attribute__((ext_vector_type(8)))  __bf16 v8bf;
typedef __attribute__((ext_vector_type(8)))  float  v8f;

#define BQ_   128
#define D_    512
#define H_    8
#define HD_   64
#define S_    3840
#define V_    12000
#define TSTEPS 15

// ---------------- WMMA fragment loaders (CDNA5 16-bit layouts) ----------------
// A: 16x32 (MxK). lane: m = lane&15, half = lane>>4.
//   elems 0..7  : k = kk + 8*half + e          (contiguous 8)
//   elems 8..15 : k = kk + 16 + 8*half + (e-8) (contiguous 8)
__device__ __forceinline__ v16bf load_frag_a(const bf16_t* __restrict__ A,
                                             int row, int K, int kk, int half) {
  const v8bf* p0 = reinterpret_cast<const v8bf*>(A + (size_t)row * K + kk + 8 * half);
  const v8bf* p1 = reinterpret_cast<const v8bf*>(A + (size_t)row * K + kk + 16 + 8 * half);
  v8bf lo = *p0, hi = *p1;
  v16bf r;
#pragma unroll
  for (int i = 0; i < 8; ++i) { r[i] = lo[i]; r[i + 8] = hi[i]; }
  return r;
}
// B: 32x16 (KxN), sourced from W row-major (N,K) so B[k][n] = W[n][k].
// lane: n = lane&15, half = lane>>4; elems 0..7: k = kk + 8*half + e; elems 8..15: k = kk + 16 + 8*half + (e-8)
__device__ __forceinline__ v16bf load_frag_b(const bf16_t* __restrict__ W,
                                             int col, int K, int kk, int half) {
  const v8bf* p0 = reinterpret_cast<const v8bf*>(W + (size_t)col * K + kk + 8 * half);
  const v8bf* p1 = reinterpret_cast<const v8bf*>(W + (size_t)col * K + kk + 16 + 8 * half);
  v8bf lo = *p0, hi = *p1;
  v16bf r;
#pragma unroll
  for (int i = 0; i < 8; ++i) { r[i] = lo[i]; r[i + 8] = hi[i]; }
  return r;
}

__device__ __forceinline__ v8f wmma_bf16(v16bf a, v16bf b, v8f c) {
  return __builtin_amdgcn_wmma_f32_16x16x32_bf16(false, a, false, b, (short)0, c, false, false);
}

// epilogue for one 16x16 C tile held in v8f (lane = col, vgpr j -> m = j + 8*half)
__device__ __forceinline__ void store_tile(v8f acc, int tm, int tn, int nn, int half,
                                           const float* __restrict__ bias,
                                           float* __restrict__ Cf, bf16_t* __restrict__ Cbf,
                                           const unsigned char* __restrict__ rowmask,
                                           int N, int accumulate) {
  const int col = tn * 16 + nn;
  const float bv = bias ? bias[col] : 0.0f;
#pragma unroll
  for (int j = 0; j < 8; ++j) {
    const int row = tm * 16 + j + 8 * half;
    float v = acc[j] + bv;
    if (accumulate) v += Cf[(size_t)row * N + col];
    if (rowmask && rowmask[row]) v = 0.0f;
    if (Cf)  Cf [(size_t)row * N + col] = v;
    if (Cbf) Cbf[(size_t)row * N + col] = (bf16_t)v;
  }
}

// ---------------- 2x2 register-blocked bf16 WMMA GEMM ----------------
// C = A(MxK) * W(N,K)^T + bias.  One wave -> 32x32 output (4 accumulators),
// 8 b128 fragment loads feed 4 WMMAs per k-chunk (2 loads/WMMA).
// Requires M%32==0, N%32==0, K%32==0 (true for every GEMM in this model).
__global__ void gemm_wmma(const bf16_t* __restrict__ A, const bf16_t* __restrict__ W,
                          const float* __restrict__ bias,
                          float* __restrict__ Cf, bf16_t* __restrict__ Cbf,
                          const unsigned char* __restrict__ rowmask,
                          int M, int N, int K, int accumulate) {
  const int lane = threadIdx.x & 31;
  const int wave = threadIdx.x >> 5;
  const int tn2  = N >> 5;                        // 32-wide supertiles in N
  const int tiles = (M >> 5) * tn2;
  const int tid  = blockIdx.x * (blockDim.x >> 5) + wave;
  if (tid >= tiles) return;                       // wave-uniform exit (EXEC all-ones inside)
  const int tm = (tid / tn2) * 2;
  const int tn = (tid % tn2) * 2;
  const int half = lane >> 4, nn = lane & 15;
  const int arow0 = tm * 16 + nn, arow1 = arow0 + 16;
  const int bcol0 = tn * 16 + nn, bcol1 = bcol0 + 16;
  v8f acc00 = {}, acc01 = {}, acc10 = {}, acc11 = {};
  for (int kk = 0; kk < K; kk += 32) {
    if (kk + 32 < K)                               // WGP-scope prefetch of next weight chunk
      __builtin_prefetch(W + (size_t)bcol0 * K + kk + 32, 0, 3);
    v16bf a0 = load_frag_a(A, arow0, K, kk, half);
    v16bf a1 = load_frag_a(A, arow1, K, kk, half);
    v16bf b0 = load_frag_b(W, bcol0, K, kk, half);
    v16bf b1 = load_frag_b(W, bcol1, K, kk, half);
    acc00 = wmma_bf16(a0, b0, acc00);
    acc01 = wmma_bf16(a0, b1, acc01);
    acc10 = wmma_bf16(a1, b0, acc10);
    acc11 = wmma_bf16(a1, b1, acc11);
  }
  store_tile(acc00, tm,     tn,     nn, half, bias, Cf, Cbf, rowmask, N, accumulate);
  store_tile(acc01, tm,     tn + 1, nn, half, bias, Cf, Cbf, rowmask, N, accumulate);
  store_tile(acc10, tm + 1, tn,     nn, half, bias, Cf, Cbf, rowmask, N, accumulate);
  store_tile(acc11, tm + 1, tn + 1, nn, half, bias, Cf, Cbf, rowmask, N, accumulate);
}

// ---------------- fused tanh-additive attention scores ----------------
// Per wave: one (bq, head) -> 16 ctx rows (K=64). Loop 32 N-tiles of ctx_W (512,64):
// c = ctx*ctx_W^T ; acc += tanh(c + ctx_b + hproj) * alpha_W ; lane-reduce -> dot (bq,h,16)
__global__ void dot_attn(const bf16_t* __restrict__ ctx_bf, const bf16_t* __restrict__ ctxW,
                         const float* __restrict__ ctx_b, const float* __restrict__ hproj,
                         const float* __restrict__ alphaW, const float* __restrict__ alphab,
                         float* __restrict__ dot) {
  const int lane = threadIdx.x & 31;
  const int wid  = blockIdx.x * (blockDim.x >> 5) + (threadIdx.x >> 5); // 0..1023 = bq*8+h
  const int bq   = wid >> 3;
  const int half = lane >> 4, nn = lane & 15;
  const int arow = wid * 16 + nn;
  const v16bf a0 = load_frag_a(ctx_bf, arow, 64, 0, half);
  const v16bf a1 = load_frag_a(ctx_bf, arow, 64, 32, half);
  float acc[8];
#pragma unroll
  for (int j = 0; j < 8; ++j) acc[j] = 0.0f;
  for (int tn = 0; tn < 32; ++tn) {
    const int col = tn * 16 + nn;
    v16bf b0 = load_frag_b(ctxW, col, 64, 0, half);
    v16bf b1 = load_frag_b(ctxW, col, 64, 32, half);
    v8f c = {};
    c = wmma_bf16(a0, b0, c);
    c = wmma_bf16(a1, b1, c);
    const float add = ctx_b[col] + hproj[bq * D_ + col];
    const float aw  = alphaW[col];
#pragma unroll
    for (int j = 0; j < 8; ++j) acc[j] += tanhf(c[j] + add) * aw;
  }
#pragma unroll
  for (int j = 0; j < 8; ++j) {
    float v = acc[j];
    v += __shfl_xor(v, 1, 32);
    v += __shfl_xor(v, 2, 32);
    v += __shfl_xor(v, 4, 32);
    v += __shfl_xor(v, 8, 32);                    // sum over the 16 lanes of this half
    if (nn == 0) dot[wid * 16 + j + 8 * half] = v + alphab[0];
  }
}

// ---------------- elementwise / small kernels ----------------
__global__ void cvt_f32_bf16(const float* __restrict__ s, bf16_t* __restrict__ d, int n) {
  int i = blockIdx.x * blockDim.x + threadIdx.x;
  if (i < n) d[i] = (bf16_t)s[i];
}

__global__ void zero_state(float* h, float* c, bf16_t* hbf) {
  int i = blockIdx.x * blockDim.x + threadIdx.x;
  if (i < BQ_ * D_) { h[i] = 0.0f; c[i] = 0.0f; hbf[i] = (bf16_t)0.0f; }
}

__global__ void build_hs_in(const bf16_t* __restrict__ hbf, const bf16_t* __restrict__ qbf,
                            bf16_t* __restrict__ hs) {
  int i = blockIdx.x * blockDim.x + threadIdx.x;          // 128*1024
  int bq = i >> 10, j = i & 1023;
  hs[i] = (j < D_) ? hbf[bq * D_ + j] : qbf[bq * D_ + (j - D_)];
}

__global__ void softmax16_k(float* __restrict__ a) {      // 1024 rows of 16
  int r = blockIdx.x * blockDim.x + threadIdx.x;
  if (r >= BQ_ * H_) return;
  float* p = a + r * 16;
  float mx = p[0];
#pragma unroll
  for (int i = 1; i < 16; ++i) mx = fmaxf(mx, p[i]);
  float e[16], s = 0.0f;
#pragma unroll
  for (int i = 0; i < 16; ++i) { e[i] = __expf(p[i] - mx); s += e[i]; }
  float inv = 1.0f / s;
#pragma unroll
  for (int i = 0; i < 16; ++i) p[i] = e[i] * inv;
}

// 1-D deformable sampling: block = one (bq, head, point), thread = feature d
__global__ void sample_ctx(const float* __restrict__ off, const float* __restrict__ aw,
                           const float* __restrict__ refp, const float* __restrict__ vratio,
                           const int* __restrict__ tshape, const int* __restrict__ lstart,
                           const bf16_t* __restrict__ value,
                           float* __restrict__ ctx_f, bf16_t* __restrict__ ctx_bf) {
  const int pt = blockIdx.x;          // (bq*8+head)*16 + pp, 16384 total
  const int d  = threadIdx.x;         // 0..63
  const int bq = pt >> 7;
  const int rem = pt & 127;           // head*16 + pp
  const int head = rem >> 4, pp = rem & 15;
  const int l = pp >> 2;
  const int b = bq >> 5;              // Q = 32
  const int Ti = tshape[l];
  const float Tl = (float)Ti;
  const int lsI = lstart[l];
  const float ref = refp[bq] * vratio[b * 4 + l];
  const float loc = ref + off[bq * 128 + rem] / Tl;
  const float x  = loc * Tl - 0.5f;
  const float x0 = floorf(x);
  const float w  = x - x0;
  const int i0 = (int)x0, i1 = i0 + 1;
  const bf16_t* vb = value + ((size_t)(b * S_ + lsI)) * D_ + head * HD_ + d;
  float g0 = 0.0f, g1 = 0.0f;
  if (i0 >= 0 && i0 < Ti) g0 = (float)vb[(size_t)i0 * D_];
  if (i1 >= 0 && i1 < Ti) g1 = (float)vb[(size_t)i1 * D_];
  const float s = g0 * (1.0f - w) + g1 * w;
  const float res = s * aw[bq * 128 + rem];
  const size_t idx = (size_t)pt * HD_ + d;
  ctx_f[idx]  = res;
  ctx_bf[idx] = (bf16_t)res;
}

// softmax over 16 points + weighted sum -> attn_res (bq, h*64+d)
__global__ void attn_reduce(const float* __restrict__ dot, const float* __restrict__ ctx_f,
                            float* __restrict__ attnres) {
  const int bqh = blockIdx.x;         // 1024
  const int d = threadIdx.x;          // 64
  const float* dp = dot + bqh * 16;
  float mx = dp[0];
#pragma unroll
  for (int p = 1; p < 16; ++p) mx = fmaxf(mx, dp[p]);
  float e[16], s = 0.0f;
#pragma unroll
  for (int p = 0; p < 16; ++p) { e[p] = __expf(dp[p] - mx); s += e[p]; }
  const float inv = 1.0f / s;
  float acc = 0.0f;
#pragma unroll
  for (int p = 0; p < 16; ++p)
    acc += e[p] * inv * ctx_f[((size_t)bqh * 16 + p) * HD_ + d];
  const int bq = bqh >> 3, h = bqh & 7;
  attnres[bq * D_ + h * HD_ + d] = acc;
}

// x = bf16 concat(embed_W[tok], attn_res, query)
__global__ void build_x(const int* __restrict__ seq, int t, const float* __restrict__ embedW,
                        const float* __restrict__ attnres, const float* __restrict__ query,
                        bf16_t* __restrict__ xbf) {
  int i = blockIdx.x * blockDim.x + threadIdx.x;          // 128*1536
  if (i >= BQ_ * 1536) return;
  int bq = i / 1536, j = i % 1536;
  float v;
  if (j < D_)       { int tok = seq[bq * 16 + t]; v = embedW[(size_t)tok * D_ + j]; }
  else if (j < 2*D_) v = attnres[bq * D_ + (j - D_)];
  else               v = query[bq * D_ + (j - 2 * D_)];
  xbf[i] = (bf16_t)v;
}

__global__ void lstm_cell(const float* __restrict__ gates, float* __restrict__ c,
                          float* __restrict__ h, bf16_t* __restrict__ hbf) {
  int i = blockIdx.x * blockDim.x + threadIdx.x;          // 128*512
  if (i >= BQ_ * D_) return;
  int bq = i >> 9, j = i & 511;
  const float* g = gates + (size_t)bq * 2048;
  const float gi = g[j], gf = g[512 + j], gg = g[1024 + j], go = g[1536 + j];
  const float si = 1.0f / (1.0f + __expf(-gi));
  const float sf = 1.0f / (1.0f + __expf(-gf));
  const float so = 1.0f / (1.0f + __expf(-go));
  const float c2 = sf * c[i] + si * tanhf(gg);
  const float h2 = so * tanhf(c2);
  c[i] = c2; h[i] = h2; hbf[i] = (bf16_t)h2;
}

__global__ void logsoftmax_row(const float* __restrict__ logits, float* __restrict__ out, int t) {
  __shared__ float red[256];
  const int bq = blockIdx.x, tid = threadIdx.x;
  const float* row = logits + (size_t)bq * V_;
  float mx = -3.0e38f;
  for (int v = tid; v < V_; v += 256) mx = fmaxf(mx, row[v]);
  red[tid] = mx; __syncthreads();
  for (int s = 128; s > 0; s >>= 1) { if (tid < s) red[tid] = fmaxf(red[tid], red[tid + s]); __syncthreads(); }
  mx = red[0]; __syncthreads();
  float sum = 0.0f;
  for (int v = tid; v < V_; v += 256) sum += __expf(row[v] - mx);
  red[tid] = sum; __syncthreads();
  for (int s = 128; s > 0; s >>= 1) { if (tid < s) red[tid] += red[tid + s]; __syncthreads(); }
  const float lse = mx + __logf(red[0]);
  float* o = out + ((size_t)bq * TSTEPS + t) * V_;
  for (int v = tid; v < V_; v += 256) o[v] = row[v] - lse;
}

// ---------------- host ----------------
extern "C" void kernel_launch(void* const* d_in, const int* in_sizes, int n_in,
                              void* d_out, int out_size, void* d_ws, size_t ws_size,
                              hipStream_t stream) {
  (void)in_sizes; (void)n_in; (void)out_size; (void)ws_size;
  const int*   seq    = (const int*)  d_in[0];
  const float* query  = (const float*)d_in[1];
  const float* refp   = (const float*)d_in[2];
  const float* enc    = (const float*)d_in[3];
  const float* vratio = (const float*)d_in[4];
  const unsigned char* mask = (const unsigned char*)d_in[5];
  const int*   tshape = (const int*)  d_in[6];
  const int*   lstart = (const int*)  d_in[7];
  const float* embedW = (const float*)d_in[8];
  const float* logitW = (const float*)d_in[9];
  const float* logitb = (const float*)d_in[10];
  const float* Wih    = (const float*)d_in[11];
  const float* Whh    = (const float*)d_in[12];
  const float* valueW = (const float*)d_in[13];
  const float* valueb = (const float*)d_in[14];
  const float* offW   = (const float*)d_in[15];
  const float* offb   = (const float*)d_in[16];
  const float* awW    = (const float*)d_in[17];
  const float* awb    = (const float*)d_in[18];
  const float* ctxW   = (const float*)d_in[19];
  const float* ctxb   = (const float*)d_in[20];
  const float* hsW    = (const float*)d_in[21];
  const float* hsb    = (const float*)d_in[22];
  const float* alphaW = (const float*)d_in[23];
  const float* alphab = (const float*)d_in[24];
  float* out = (float*)d_out;

  char* base = (char*)d_ws;
  size_t woff = 0;
  auto alloc = [&](size_t bytes) -> void* {
    size_t o = (woff + 255) & ~(size_t)255;
    woff = o + bytes;
    return (void*)(base + o);
  };
  // bf16 weight copies (fit in 192MB L2; halve bandwidth vs fp32)
  bf16_t* enc_bf    = (bf16_t*)alloc((size_t)4 * S_ * D_ * 2);
  bf16_t* valW_bf   = (bf16_t*)alloc((size_t)D_ * D_ * 2);
  bf16_t* offW_bf   = (bf16_t*)alloc((size_t)128 * 1024 * 2);
  bf16_t* awW_bf    = (bf16_t*)alloc((size_t)128 * 1024 * 2);
  bf16_t* hsW_bf    = (bf16_t*)alloc((size_t)D_ * D_ * 2);
  bf16_t* ctxW_bf   = (bf16_t*)alloc((size_t)D_ * HD_ * 2);
  bf16_t* Wih_bf    = (bf16_t*)alloc((size_t)2048 * 1536 * 2);
  bf16_t* Whh_bf    = (bf16_t*)alloc((size_t)2048 * 512 * 2);
  bf16_t* logitW_bf = (bf16_t*)alloc((size_t)V_ * D_ * 2);
  bf16_t* query_bf  = (bf16_t*)alloc((size_t)BQ_ * D_ * 2);
  // state / scratch
  bf16_t* value_bf  = (bf16_t*)alloc((size_t)4 * S_ * D_ * 2);
  float*  h_f       = (float*) alloc((size_t)BQ_ * D_ * 4);
  float*  c_f       = (float*) alloc((size_t)BQ_ * D_ * 4);
  bf16_t* h_bf      = (bf16_t*)alloc((size_t)BQ_ * D_ * 2);
  bf16_t* hs_in_bf  = (bf16_t*)alloc((size_t)BQ_ * 1024 * 2);
  float*  off_f     = (float*) alloc((size_t)BQ_ * 128 * 4);
  float*  aw_f      = (float*) alloc((size_t)BQ_ * 128 * 4);
  float*  hproj_f   = (float*) alloc((size_t)BQ_ * D_ * 4);
  float*  ctx_f     = (float*) alloc((size_t)BQ_ * H_ * 16 * HD_ * 4);
  bf16_t* ctx_bf    = (bf16_t*)alloc((size_t)BQ_ * H_ * 16 * HD_ * 2);
  float*  dot_f     = (float*) alloc((size_t)BQ_ * H_ * 16 * 4);
  float*  attnres_f = (float*) alloc((size_t)BQ_ * D_ * 4);
  bf16_t* x_bf      = (bf16_t*)alloc((size_t)BQ_ * 1536 * 2);
  float*  gates_f   = (float*) alloc((size_t)BQ_ * 2048 * 4);
  float*  logits_f  = (float*) alloc((size_t)BQ_ * V_ * 4);

  auto cvt = [&](const float* s, bf16_t* d, int n) {
    cvt_f32_bf16<<<(n + 255) / 256, 256, 0, stream>>>(s, d, n);
  };
  auto gemm = [&](const bf16_t* A, const bf16_t* W, const float* bias,
                  float* Cf, bf16_t* Cbf, const unsigned char* rowmask,
                  int M, int N, int K, int accum) {
    int tiles = (M / 32) * (N / 32);                   // one wave per 32x32 supertile
    gemm_wmma<<<(tiles + 7) / 8, 256, 0, stream>>>(A, W, bias, Cf, Cbf, rowmask, M, N, K, accum);
  };

  // one-time conversions
  cvt(enc,    enc_bf,    4 * S_ * D_);
  cvt(valueW, valW_bf,   D_ * D_);
  cvt(offW,   offW_bf,   128 * 1024);
  cvt(awW,    awW_bf,    128 * 1024);
  cvt(hsW,    hsW_bf,    D_ * D_);
  cvt(ctxW,   ctxW_bf,   D_ * HD_);
  cvt(Wih,    Wih_bf,    2048 * 1536);
  cvt(Whh,    Whh_bf,    2048 * 512);
  cvt(logitW, logitW_bf, V_ * D_);
  cvt(query,  query_bf,  BQ_ * D_);

  zero_state<<<256, 256, 0, stream>>>(h_f, c_f, h_bf);
  // value = mask ? 0 : enc @ value_W^T + value_b   (stored bf16, layout (b,s,h*hd))
  gemm(enc_bf, valW_bf, valueb, nullptr, value_bf, mask, 4 * S_, D_, D_, 0);

  for (int t = 0; t < TSTEPS; ++t) {
    build_hs_in<<<(BQ_ * 1024) / 256, 256, 0, stream>>>(h_bf, query_bf, hs_in_bf);
    gemm(hs_in_bf, offW_bf, offb, off_f, nullptr, nullptr, BQ_, 128, 1024, 0);
    gemm(hs_in_bf, awW_bf,  awb,  aw_f,  nullptr, nullptr, BQ_, 128, 1024, 0);
    softmax16_k<<<4, 256, 0, stream>>>(aw_f);
    gemm(h_bf, hsW_bf, hsb, hproj_f, nullptr, nullptr, BQ_, D_, D_, 0);
    sample_ctx<<<BQ_ * H_ * 16, HD_, 0, stream>>>(off_f, aw_f, refp, vratio,
                                                  tshape, lstart, value_bf, ctx_f, ctx_bf);
    dot_attn<<<128, 256, 0, stream>>>(ctx_bf, ctxW_bf, ctxb, hproj_f, alphaW, alphab, dot_f);
    attn_reduce<<<BQ_ * H_, HD_, 0, stream>>>(dot_f, ctx_f, attnres_f);
    build_x<<<(BQ_ * 1536) / 256, 256, 0, stream>>>(seq, t, embedW, attnres_f, query, x_bf);
    gemm(x_bf, Wih_bf, nullptr, gates_f, nullptr, nullptr, BQ_, 2048, 1536, 0);
    gemm(h_bf, Whh_bf, nullptr, gates_f, nullptr, nullptr, BQ_, 2048, 512, 1);
    lstm_cell<<<256, 256, 0, stream>>>(gates_f, c_f, h_f, h_bf);
    gemm(h_bf, logitW_bf, logitb, logits_f, nullptr, nullptr, BQ_, V_, 512, 0);
    logsoftmax_row<<<BQ_, 256, 0, stream>>>(logits_f, out, t);
  }
}
